// MultiScaleMaskedTransformerDecoder_50577534878488
// MI455X (gfx1250) — matrix operand order repaired and checked by reading, not compiled, not run on hardware
//
#include <hip/hip_runtime.h>
#include <hip/hip_bf16.h>

// ---------------- problem constants ----------------
#define D_MODEL 256
#define NHEAD   8
#define HEADD   32
#define NEXP    5
#define LQ      100
#define LK      1024
#define BATCH   16
#define LQPAD   112              // 7 tiles of 16
#define ROWS_Q  (LQ * BATCH)     // 1600
#define ROWS_K  (LK * BATCH)     // 16384
#define QSCALE  0.17677669529663687f   // 1/sqrt(32)
#define LN_EPS  1e-5f

typedef __bf16 v16bf __attribute__((ext_vector_type(16)));
typedef float  v8f   __attribute__((ext_vector_type(8)));
typedef unsigned short u16;

union BFrag {                    // 16 bf16 = 32B = one WMMA operand per lane
    v16bf v;
    uint4 q[2];
    u16   us[16];
};

__device__ __forceinline__ u16 f2bf(float f) {
    union { float f; unsigned u; } c; c.f = f;
    unsigned u = c.u;
    u += 0x7FFFu + ((u >> 16) & 1u);     // round-to-nearest-even
    return (u16)(u >> 16);
}

__device__ __forceinline__ v8f wmma_bf16(const BFrag& a, const BFrag& b, v8f c) {
    return __builtin_amdgcn_wmma_f32_16x16x32_bf16(false, a.v, false, b.v,
                                                   (short)0, c, false, false);
}

// ---------------- prep: fp32 -> bf16 packing ----------------
__global__ void prep_kernel(const float* __restrict__ tgt, const float* __restrict__ memory,
                            const float* __restrict__ qp,  const float* __restrict__ pos,
                            const float* __restrict__ w_in, const float* __restrict__ w_out,
                            u16* __restrict__ qemb16, u16* __restrict__ kemb16,
                            u16* __restrict__ mem16,  u16* __restrict__ w_in16,
                            u16* __restrict__ w_out16) {
    const int NQ  = ROWS_Q * D_MODEL;          // 409600
    const int NK  = ROWS_K * D_MODEL;          // 4194304
    const int NWI = NEXP * 3 * D_MODEL * D_MODEL;
    const int NWO = NEXP * D_MODEL * D_MODEL;
    const int total = NQ + NK + NK + NWI + NWO;
    for (int i = blockIdx.x * blockDim.x + threadIdx.x; i < total;
         i += gridDim.x * blockDim.x) {
        int j = i;
        if (j < NQ)  { qemb16[j] = f2bf(tgt[j] + qp[j]);     continue; }
        j -= NQ;
        if (j < NK)  { kemb16[j] = f2bf(memory[j] + pos[j]); continue; }
        j -= NK;
        if (j < NK)  { mem16[j]  = f2bf(memory[j]);          continue; }
        j -= NK;
        if (j < NWI) { w_in16[j] = f2bf(w_in[j]);            continue; }
        j -= NWI;
        w_out16[j] = f2bf(w_out[j]);
    }
}

// ---------------- gate softmax: one wave per (l,b) row ----------------
__global__ void gate_kernel(const float* __restrict__ tgt, const float* __restrict__ w_gate,
                            const float* __restrict__ b_gate, float* __restrict__ gate) {
    int wid  = (blockIdx.x * blockDim.x + threadIdx.x) >> 5;
    int lane = threadIdx.x & 31;
    if (wid >= ROWS_Q) return;
    float acc[NEXP] = {0.f, 0.f, 0.f, 0.f, 0.f};
    #pragma unroll
    for (int j = 0; j < 8; ++j) {
        float x = tgt[wid * D_MODEL + lane + 32 * j];
        #pragma unroll
        for (int e = 0; e < NEXP; ++e)
            acc[e] += x * w_gate[e * D_MODEL + lane + 32 * j];
    }
    #pragma unroll
    for (int e = 0; e < NEXP; ++e)
        for (int off = 16; off > 0; off >>= 1)
            acc[e] += __shfl_xor(acc[e], off, 32);
    float v[NEXP], mx = -3e38f;
    #pragma unroll
    for (int e = 0; e < NEXP; ++e) { v[e] = acc[e] + b_gate[e]; mx = fmaxf(mx, v[e]); }
    float s = 0.f;
    #pragma unroll
    for (int e = 0; e < NEXP; ++e) { v[e] = __expf(v[e] - mx); s += v[e]; }
    if (lane == 0) {
        float inv = 1.f / s;
        #pragma unroll
        for (int e = 0; e < NEXP; ++e) gate[wid * NEXP + e] = v[e] * inv;
    }
}

// ---------------- Q/K/V projection GEMM (TN, bf16 WMMA, 2x2 register blocking) ----
// C[m,o] = sum_k A[m,k]*W[o,k]; mode 0: Q (scale+scatter), 1: K, 2: V^T
// Each wave computes a 32x32 output tile: 4 accumulators, 2 A-frags + 2 B-frags
// per k-step -> 4 WMMAs per 4 operand fetches (2x the operand reuse of 1 tile/wave).
__global__ void proj_kernel(const u16* __restrict__ A, const u16* __restrict__ W,
                            const float* __restrict__ bias, u16* __restrict__ dst,
                            int mode, int Mtiles32) {
    int wave = (blockIdx.x * blockDim.x + threadIdx.x) >> 5;
    int lane = threadIdx.x & 31;
    int tileN2 = wave & 7;            // 8 tiles of 32 over N=256
    int tileM2 = wave >> 3;
    if (tileM2 >= Mtiles32) return;
    int half = lane >> 4;
    int l15  = lane & 15;
    int kaA  = half * 16;    // A-layout: lanes 0-15 K 0..15, 16-31 K 16..31
    int kaB  = half * 16;    // B-layout: lanes 0-15 K 0..15, 16-31 K 16..31
    const uint4* Arow0 = (const uint4*)(A + (tileM2 * 32 + l15) * D_MODEL);
    const uint4* Arow1 = (const uint4*)(A + (tileM2 * 32 + 16 + l15) * D_MODEL);
    const uint4* Wrow0 = (const uint4*)(W + (tileN2 * 32 + l15) * D_MODEL);
    const uint4* Wrow1 = (const uint4*)(W + (tileN2 * 32 + 16 + l15) * D_MODEL);
    v8f acc00 = {}, acc01 = {}, acc10 = {}, acc11 = {};
    #pragma unroll
    for (int k = 0; k < D_MODEL; k += 32) {
        BFrag a0, a1, b0, b1;
        a0.q[0] = Arow0[(k + kaA) >> 3];        a0.q[1] = Arow0[(k + kaA + 8) >> 3];
        a1.q[0] = Arow1[(k + kaA) >> 3];        a1.q[1] = Arow1[(k + kaA + 8) >> 3];
        b0.q[0] = Wrow0[(k + kaB) >> 3];        b0.q[1] = Wrow0[((k + kaB) >> 3) + 1];
        b1.q[0] = Wrow1[(k + kaB) >> 3];        b1.q[1] = Wrow1[((k + kaB) >> 3) + 1];
        acc00 = wmma_bf16(a0, b0, acc00);
        acc01 = wmma_bf16(a0, b1, acc01);
        acc10 = wmma_bf16(a1, b0, acc10);
        acc11 = wmma_bf16(a1, b1, acc11);
    }
    v8f accs[2][2] = {{acc00, acc01}, {acc10, acc11}};
    #pragma unroll
    for (int mi = 0; mi < 2; ++mi) {
        #pragma unroll
        for (int ni = 0; ni < 2; ++ni) {
            v8f acc = accs[mi][ni];
            int o = tileN2 * 32 + ni * 16 + l15;
            int h = o >> 5, kk = o & 31;
            float bv = bias[o];
            #pragma unroll
            for (int r = 0; r < 8; ++r) {
                int m = tileM2 * 32 + mi * 16 + r + 8 * half;  // D-layout: M across VGPRs
                float val = acc[r] + bv;
                int bb = m & 15;
                if (mode == 0) {            // Q[b][h][l][kk], pre-scaled
                    int l = m >> 4;
                    dst[(((bb << 3) + h) * LQPAD + l) * HEADD + kk] = f2bf(val * QSCALE);
                } else if (mode == 1) {     // K[b][h][s][kk]
                    int s = m >> 4;
                    dst[(((bb << 3) + h) * LK + s) * HEADD + kk] = f2bf(val);
                } else {                    // V^T[b][h][kk][s]  (s contiguous)
                    int s = m >> 4;
                    dst[(((bb << 3) + h) * HEADD + kk) * LK + s] = f2bf(val);
                }
            }
        }
    }
}

// ---------------- fused flash attention: one wave per (b,h,ltile) ----------------
__global__ void attn_kernel(const u16* __restrict__ Qe, const u16* __restrict__ Ke,
                            const u16* __restrict__ Vte, u16* __restrict__ ctx16) {
    __shared__ __align__(16) u16 ldsP[4][16][32];
    int lane = threadIdx.x & 31;
    int w    = threadIdx.x >> 5;
    int bh   = blockIdx.x >> 1;
    int lt   = (blockIdx.x & 1) * 4 + w;
    if (lt >= 7) return;                       // wave-uniform; EXEC stays all-ones
    int b = bh >> 3, h = bh & 7;
    int half = lane >> 4;
    int kaA  = half * 8;
    // Q fragment (rows l = lt*16.., K = head dim 32): loaded once
    const u16* Qb = Qe + (bh * LQPAD + lt * 16 + (lane & 15)) * HEADD;
    BFrag qa;
    qa.q[0] = *(const uint4*)(Qb + kaA);
    qa.q[1] = *(const uint4*)(Qb + kaA + 16);
    const u16* Kb = Ke  + (size_t)bh * LK * HEADD;
    const u16* Vb = Vte + (size_t)bh * HEADD * LK;
    v8f ctx0 = {}, ctx1 = {};
    float rmax[8], rsum[8];
    #pragma unroll
    for (int r = 0; r < 8; ++r) { rmax[r] = -3e38f; rsum[r] = 0.f; }

    for (int s0 = 0; s0 < LK; s0 += 32) {
        // K fragments as B operand: B[k][n] = K[s0 + tile*16 + n][k]
        BFrag kb0, kb1;
        const u16* kp0 = Kb + (s0 + (lane & 15)) * HEADD + half * 8;
        const u16* kp1 = Kb + (s0 + 16 + (lane & 15)) * HEADD + half * 8;
        kb0.q[0] = *(const uint4*)kp0;  kb0.q[1] = *(const uint4*)(kp0 + 16);
        kb1.q[0] = *(const uint4*)kp1;  kb1.q[1] = *(const uint4*)(kp1 + 16);
        v8f sa = {}, sb = {};
        sa = wmma_bf16(qa, kb0, sa);    // scores[l][s0..s0+15]
        sb = wmma_bf16(qa, kb1, sb);    // scores[l][s0+16..s0+31]
        // online softmax per row (rows striped across VGPRs, N across lanes)
        #pragma unroll
        for (int r = 0; r < 8; ++r) {
            float cm = fmaxf(sa[r], sb[r]);
            #pragma unroll
            for (int off = 1; off < 16; off <<= 1) cm = fmaxf(cm, __shfl_xor(cm, off, 32));
            float mo = rmax[r], mn = fmaxf(mo, cm);
            float corr = __expf(mo - mn);
            rmax[r] = mn;
            float p0 = __expf(sa[r] - mn), p1 = __expf(sb[r] - mn);
            float rs = p0 + p1;
            #pragma unroll
            for (int off = 1; off < 16; off <<= 1) rs += __shfl_xor(rs, off, 32);
            rsum[r] = rsum[r] * corr + rs;
            ctx0[r] *= corr;  ctx1[r] *= corr;
            int m = r + 8 * half, n = lane & 15;     // C/D layout -> [row][col]
            ldsP[w][m][n]      = f2bf(p0);
            ldsP[w][m][n + 16] = f2bf(p1);
        }
        asm volatile("s_wait_dscnt 0" ::: "memory");
        // reload P in A-operand layout from LDS
        BFrag pa;
        const u16* pr = &ldsP[w][lane & 15][kaA];
        pa.q[0] = *(const uint4*)pr;
        pa.q[1] = *(const uint4*)(pr + 16);
        // V fragments: B[k][n] = Vt[kk = tile*16 + n][s0 + k]
        BFrag vb0, vb1;
        const u16* vp0 = Vb + ((lane & 15)) * LK + s0 + half * 8;
        const u16* vp1 = Vb + (16 + (lane & 15)) * LK + s0 + half * 8;
        vb0.q[0] = *(const uint4*)vp0;  vb0.q[1] = *(const uint4*)(vp0 + 16);
        vb1.q[0] = *(const uint4*)vp1;  vb1.q[1] = *(const uint4*)(vp1 + 16);
        ctx0 = wmma_bf16(pa, vb0, ctx0);   // ctx[l][kk 0..15]
        ctx1 = wmma_bf16(pa, vb1, ctx1);   // ctx[l][kk 16..31]
    }
    #pragma unroll
    for (int r = 0; r < 8; ++r) {
        int l = lt * 16 + r + 8 * half;
        if (l < LQ) {
            float inv = 1.f / rsum[r];
            int n = lane & 15;
            size_t row = (size_t)(l * BATCH + b) * D_MODEL + h * HEADD;
            ctx16[row + n]      = f2bf(ctx0[r] * inv);
            ctx16[row + 16 + n] = f2bf(ctx1[r] * inv);
        }
    }
}

// ---------------- output projection + gated accumulation (2x2 blocking) --------
__global__ void outproj_kernel(const u16* __restrict__ A, const u16* __restrict__ W,
                               const float* __restrict__ bias, const float* __restrict__ gate,
                               float* __restrict__ moacc, int e) {
    int wave = (blockIdx.x * blockDim.x + threadIdx.x) >> 5;
    int lane = threadIdx.x & 31;
    int tileN2 = wave & 7;
    int tileM2 = wave >> 3;
    if (tileM2 >= ROWS_Q / 32) return;
    int half = lane >> 4;
    int l15  = lane & 15;
    int kaA  = half * 16, kaB = half * 16;
    const uint4* Arow0 = (const uint4*)(A + (tileM2 * 32 + l15) * D_MODEL);
    const uint4* Arow1 = (const uint4*)(A + (tileM2 * 32 + 16 + l15) * D_MODEL);
    const uint4* Wrow0 = (const uint4*)(W + (tileN2 * 32 + l15) * D_MODEL);
    const uint4* Wrow1 = (const uint4*)(W + (tileN2 * 32 + 16 + l15) * D_MODEL);
    v8f acc00 = {}, acc01 = {}, acc10 = {}, acc11 = {};
    #pragma unroll
    for (int k = 0; k < D_MODEL; k += 32) {
        BFrag a0, a1, b0, b1;
        a0.q[0] = Arow0[(k + kaA) >> 3];        a0.q[1] = Arow0[(k + kaA + 8) >> 3];
        a1.q[0] = Arow1[(k + kaA) >> 3];        a1.q[1] = Arow1[(k + kaA + 8) >> 3];
        b0.q[0] = Wrow0[(k + kaB) >> 3];        b0.q[1] = Wrow0[((k + kaB) >> 3) + 1];
        b1.q[0] = Wrow1[(k + kaB) >> 3];        b1.q[1] = Wrow1[((k + kaB) >> 3) + 1];
        acc00 = wmma_bf16(a0, b0, acc00);
        acc01 = wmma_bf16(a0, b1, acc01);
        acc10 = wmma_bf16(a1, b0, acc10);
        acc11 = wmma_bf16(a1, b1, acc11);
    }
    v8f accs[2][2] = {{acc00, acc01}, {acc10, acc11}};
    #pragma unroll
    for (int mi = 0; mi < 2; ++mi) {
        #pragma unroll
        for (int ni = 0; ni < 2; ++ni) {
            v8f acc = accs[mi][ni];
            int o = tileN2 * 32 + ni * 16 + l15;
            float bv = bias[o];
            #pragma unroll
            for (int r = 0; r < 8; ++r) {
                int m = tileM2 * 32 + mi * 16 + r + 8 * half;
                float val = acc[r] + bv;
                moacc[m * D_MODEL + o] += gate[m * NEXP + e] * val; // one lane owns (m,o)
            }
        }
    }
}

// ---------------- residual + LayerNorm: one wave per row ----------------
__global__ void final_kernel(const float* __restrict__ tgt, const float* __restrict__ moacc,
                             const float* __restrict__ gamma, const float* __restrict__ beta,
                             float* __restrict__ out) {
    int wid  = (blockIdx.x * blockDim.x + threadIdx.x) >> 5;
    int lane = threadIdx.x & 31;
    if (wid >= ROWS_Q) return;
    float x[8], s = 0.f, s2 = 0.f;
    #pragma unroll
    for (int j = 0; j < 8; ++j) {
        int c = lane + 32 * j;
        x[j] = tgt[wid * D_MODEL + c] + moacc[wid * D_MODEL + c];
        s += x[j];  s2 += x[j] * x[j];
    }
    #pragma unroll
    for (int off = 16; off > 0; off >>= 1) {
        s  += __shfl_xor(s, off, 32);
        s2 += __shfl_xor(s2, off, 32);
    }
    float mu  = s * (1.f / D_MODEL);
    float var = s2 * (1.f / D_MODEL) - mu * mu;
    float inv = rsqrtf(var + LN_EPS);
    #pragma unroll
    for (int j = 0; j < 8; ++j) {
        int c = lane + 32 * j;
        out[wid * D_MODEL + c] = (x[j] - mu) * inv * gamma[c] + beta[c];
    }
}

// ---------------- launch ----------------
extern "C" void kernel_launch(void* const* d_in, const int* in_sizes, int n_in,
                              void* d_out, int out_size, void* d_ws, size_t ws_size,
                              hipStream_t stream) {
    const float* tgt    = (const float*)d_in[0];
    const float* memory = (const float*)d_in[1];
    const float* qp     = (const float*)d_in[2];
    const float* pos    = (const float*)d_in[3];
    const float* w_in   = (const float*)d_in[4];
    const float* b_in   = (const float*)d_in[5];
    const float* w_out  = (const float*)d_in[6];
    const float* b_out  = (const float*)d_in[7];
    const float* w_gate = (const float*)d_in[8];
    const float* b_gate = (const float*)d_in[9];
    const float* gamma  = (const float*)d_in[10];
    const float* beta   = (const float*)d_in[11];
    float* out = (float*)d_out;

    char* ws = (char*)d_ws;
    size_t off = 0;
    auto alloc = [&](size_t bytes) { size_t o = off; off = (off + bytes + 255) & ~(size_t)255; return o; };
    float* gateBuf = (float*)(ws + alloc((size_t)ROWS_Q * NEXP * 4));
    float* moBuf   = (float*)(ws + alloc((size_t)ROWS_Q * D_MODEL * 4));
    u16* qemb16    = (u16*)(ws + alloc((size_t)ROWS_Q * D_MODEL * 2));
    u16* kemb16    = (u16*)(ws + alloc((size_t)ROWS_K * D_MODEL * 2));
    u16* mem16     = (u16*)(ws + alloc((size_t)ROWS_K * D_MODEL * 2));
    u16* w_in16    = (u16*)(ws + alloc((size_t)NEXP * 3 * D_MODEL * D_MODEL * 2));
    u16* w_out16   = (u16*)(ws + alloc((size_t)NEXP * D_MODEL * D_MODEL * 2));
    size_t QeBytes = (size_t)BATCH * NHEAD * LQPAD * HEADD * 2;
    u16* Qe        = (u16*)(ws + alloc(QeBytes));
    u16* Ke        = (u16*)(ws + alloc((size_t)BATCH * NHEAD * LK * HEADD * 2));
    u16* Vte       = (u16*)(ws + alloc((size_t)BATCH * NHEAD * HEADD * LK * 2));
    u16* ctx16     = (u16*)(ws + alloc((size_t)ROWS_Q * D_MODEL * 2));

    hipMemsetAsync(moBuf, 0, (size_t)ROWS_Q * D_MODEL * 4, stream);
    prep_kernel<<<2048, 256, 0, stream>>>(tgt, memory, qp, pos, w_in, w_out,
                                          qemb16, kemb16, mem16, w_in16, w_out16);
    gate_kernel<<<200, 256, 0, stream>>>(tgt, w_gate, b_gate, gateBuf);

    for (int e = 0; e < NEXP; ++e) {
        const u16* wi = w_in16 + (size_t)e * 3 * D_MODEL * D_MODEL;
        const float* bi = b_in + (size_t)e * 3 * D_MODEL;
        hipMemsetAsync(Qe, 0, QeBytes, stream);   // zero l >= 100 padding rows
        // waves per proj = (M/32) * 8; 8 waves per 256-thread block
        proj_kernel<<<(ROWS_Q / 32), 256, 0, stream>>>(qemb16, wi, bi, Qe, 0, ROWS_Q / 32);
        proj_kernel<<<(ROWS_K / 32), 256, 0, stream>>>(kemb16, wi + D_MODEL * D_MODEL,
                                                       bi + D_MODEL, Ke, 1, ROWS_K / 32);
        proj_kernel<<<(ROWS_K / 32), 256, 0, stream>>>(mem16, wi + 2 * D_MODEL * D_MODEL,
                                                       bi + 2 * D_MODEL, Vte, 2, ROWS_K / 32);
        attn_kernel<<<BATCH * NHEAD * 2, 128, 0, stream>>>(Qe, Ke, Vte, ctx16);
        outproj_kernel<<<(ROWS_Q / 32), 256, 0, stream>>>(ctx16,
                                                w_out16 + (size_t)e * D_MODEL * D_MODEL,
                                                b_out + (size_t)e * D_MODEL, gateBuf, moBuf, e);
    }
    final_kernel<<<200, 256, 0, stream>>>(tgt, moBuf, gamma, beta, out);
}
